// SpikeGraphTransformer_89670327206079
// MI455X (gfx1250) — matrix operand order repaired and checked by reading, not compiled, not run on hardware
//
#include <hip/hip_runtime.h>

typedef __attribute__((ext_vector_type(16))) _Float16 v16h;
typedef __attribute__((ext_vector_type(8)))  _Float16 v8h;
typedef __attribute__((ext_vector_type(8)))  float    v8f;

#define N_NODES 16384
#define FDIM    512
#define DDIM    512
#define DHDIM   1024
#define CDIM    40
#define TT      2
#define HHEADS  8
#define LLAYERS 4
#define EEDGES  262144
#define EPSV    1e-5f

// ---------------------------------------------------------------------------
// CDNA5 async global->LDS copy (ASYNCcnt path), guarded so the file compiles
// on toolchains without the builtin (falls back to register staging).
// Builtin signature (from clang diagnostic): param0 = addrspace(1)
// vector_size(16) int* (global src), param1 = LDS dst, then imm offset, cpol.
// ---------------------------------------------------------------------------
#if defined(__has_builtin)
#if __has_builtin(__builtin_amdgcn_global_load_async_to_lds_b128)
#define SGT_ASYNC 1
#endif
#endif

#ifdef SGT_ASYNC
typedef int sgt_v4i __attribute__((__vector_size__(16)));
static __device__ inline void sgt_cp_async_b128(const _Float16* g, _Float16* l) {
  __builtin_amdgcn_global_load_async_to_lds_b128(
      (__attribute__((address_space(1))) sgt_v4i*)(unsigned long long)(g),
      (__attribute__((address_space(3))) sgt_v4i*)(unsigned int)(unsigned long long)(l),
      0, 0);
}
static __device__ inline void sgt_wait_async() {
#if __has_builtin(__builtin_amdgcn_s_wait_asynccnt)
  __builtin_amdgcn_s_wait_asynccnt(0);
#else
  asm volatile("s_wait_asynccnt 0" ::: "memory");
#endif
}
#endif

// ---------------------------------------------------------------------------
// f32 -> f16 conversion
// ---------------------------------------------------------------------------
__global__ void sgt_cvt_f16(const float* __restrict__ s, _Float16* __restrict__ d, int n) {
  int i = blockIdx.x * blockDim.x + threadIdx.x;
  int st = gridDim.x * blockDim.x;
  for (; i < n; i += st) d[i] = (_Float16)s[i];
}

// ---------------------------------------------------------------------------
// WMMA GEMM: C[M,Nc] = A[M,K](f16) * B[K,Nc](f16) + bias, f32 accumulate.
// Block tile 128x128, 8 waves (4x2), wave tile 32x64 = 2x4 WMMA 16x16x32 f16.
// Double-buffered LDS, one barrier per K-step. A tile staged via async DMA
// (when available), B tile register-prefetched and stored transposed as
// ds_store_b32 k-pairs. global_prefetch for tile k+2.
// ---------------------------------------------------------------------------
#define GBM 128
#define GBN 128
#define GBK 32
#define APAD 40   // halfs per LDS A row (80B rows, 16B-aligned)
#define BPAD 40   // halfs per LDS B row (B stored transposed: Bs[c][k])

// Load 8 halfs of two consecutive B k-rows (columns nBase+c0 .. +7).
static __device__ inline void sgt_loadBpair(const _Float16* __restrict__ B, int krow,
                                            int Nc, int nBase, int c0,
                                            v8h& r0, v8h& r1) {
  const _Float16* bp0 = B + (size_t)krow * Nc + nBase + c0;
  const _Float16* bp1 = bp0 + Nc;
  if (nBase + c0 + 8 <= Nc) {
    r0 = *(const v8h*)(bp0);
    r1 = *(const v8h*)(bp1);
  } else {
    #pragma unroll
    for (int j = 0; j < 8; ++j) {
      int c = nBase + c0 + j;
      r0[j] = (c < Nc) ? bp0[j] : (_Float16)0.0f;
      r1[j] = (c < Nc) ? bp1[j] : (_Float16)0.0f;
    }
  }
}

// Store 8 transposed k-pairs: Bs[c0+j][kp], Bs[c0+j][kp+1] as one b32 each.
static __device__ inline void sgt_storeBpair(_Float16* __restrict__ base, int c0, int kp,
                                             v8h r0, v8h r1) {
  #pragma unroll
  for (int j = 0; j < 8; ++j) {
    union { _Float16 h[2]; unsigned int u; } pk;
    pk.h[0] = r0[j];
    pk.h[1] = r1[j];
    *(unsigned int*)((void*)&base[(c0 + j) * BPAD + kp]) = pk.u;
  }
}

__global__ __launch_bounds__(256)
void sgt_gemm(const _Float16* __restrict__ A, const _Float16* __restrict__ B,
              float* __restrict__ Cmat, const float* __restrict__ bias,
              int M, int K, int Nc) {
  __shared__ _Float16 As[2][GBM * APAD];   // 2 x 10 KB
  __shared__ _Float16 Bs[2][GBN * BPAD];   // 2 x 10 KB
  const int tid  = threadIdx.x;
  const int wave = tid >> 5, lane = tid & 31;
  const int lr = lane & 15, hi = lane >> 4;
  const int rowG = wave >> 1, colG = wave & 1;   // 4x2 wave grid, tile 32x64
  const int mBase = blockIdx.x * GBM;
  const int nBase = blockIdx.y * GBN;

  const v8f vz = {0.f,0.f,0.f,0.f,0.f,0.f,0.f,0.f};
  v8f acc[2][4];
  #pragma unroll
  for (int i = 0; i < 2; ++i)
    #pragma unroll
    for (int j = 0; j < 4; ++j) acc[i][j] = vz;

  // A staging: each thread owns 16 halfs of one A row per K-step
  const int aRow = tid >> 1;
  const int aCol = (tid & 1) * 16;
  // B staging: each thread owns two k-rows x 8 columns per K-step
  const int bC8 = (tid & 15) * 8;    // 0..120
  const int bKp = (tid >> 4) * 2;    // 0..30

  const _Float16* aPtr = A + (size_t)(mBase + aRow) * K + aCol;

  // ---- preload tile 0 into buffer 0 ----
  {
#ifdef SGT_ASYNC
    _Float16* ad = &As[0][aRow * APAD + aCol];
    sgt_cp_async_b128(aPtr, ad);
    sgt_cp_async_b128(aPtr + 8, ad + 8);
#else
    v8h ia0 = *(const v8h*)(aPtr);
    v8h ia1 = *(const v8h*)(aPtr + 8);
    _Float16* ad = &As[0][aRow * APAD + aCol];
    *(v8h*)(ad)     = ia0;
    *(v8h*)(ad + 8) = ia1;
#endif
    v8h ib0, ib1;
    sgt_loadBpair(B, bKp, Nc, nBase, bC8, ib0, ib1);
    sgt_storeBpair(&Bs[0][0], bC8, bKp, ib0, ib1);
#ifdef SGT_ASYNC
    sgt_wait_async();
#endif
  }
  __syncthreads();

  int cur = 0;
  for (int k0 = 0; k0 < K; k0 += GBK) {
    const bool have = (k0 + GBK) < K;      // uniform

    // prefetch next tile (A via async DMA, B into registers)
    v8h nb0, nb1;
    if (have) {
#ifdef SGT_ASYNC
      const _Float16* ap = aPtr + (k0 + GBK);
      _Float16* ad = &As[cur ^ 1][aRow * APAD + aCol];
      sgt_cp_async_b128(ap, ad);
      sgt_cp_async_b128(ap + 8, ad + 8);
#endif
      sgt_loadBpair(B, k0 + GBK + bKp, Nc, nBase, bC8, nb0, nb1);
      if (k0 + 2 * GBK < K) {   // speculative prefetch of tile k+2
        __builtin_prefetch(aPtr + (k0 + 2 * GBK), 0, 0);
        __builtin_prefetch(B + (size_t)(k0 + 2 * GBK + bKp) * Nc + nBase + bC8, 0, 0);
      }
    }
#ifndef SGT_ASYNC
    v8h na0, na1;
    if (have) {
      const _Float16* ap = aPtr + (k0 + GBK);
      na0 = *(const v8h*)(ap);
      na1 = *(const v8h*)(ap + 8);
    }
#endif

    // fragments from LDS per ISA 16-bit wave32 layouts
    v16h afr[2], bfr[4];
    #pragma unroll
    for (int i = 0; i < 2; ++i) {
      const _Float16* ar = &As[cur][(rowG * 32 + i * 16 + lr) * APAD];
      v8h lo = *(const v8h*)(ar + hi * 8);        // K = hi*8 .. +7
      v8h hb = *(const v8h*)(ar + 16 + hi * 8);   // K = 16+hi*8 .. +7
      afr[i] = __builtin_shufflevector(lo, hb, 0,1,2,3,4,5,6,7,8,9,10,11,12,13,14,15);
    }
    #pragma unroll
    for (int j = 0; j < 4; ++j) {
      const _Float16* br = &Bs[cur][(colG * 64 + j * 16 + lr) * BPAD + hi * 8];
      v8h lo = *(const v8h*)(br);                 // K = hi*8 .. +7
      v8h hb = *(const v8h*)(br + 16);            // K = 16+hi*8 .. +7
      bfr[j] = __builtin_shufflevector(lo, hb, 0,1,2,3,4,5,6,7,8,9,10,11,12,13,14,15);
    }
    #pragma unroll
    for (int i = 0; i < 2; ++i)
      #pragma unroll
      for (int j = 0; j < 4; ++j)
        acc[i][j] = __builtin_amdgcn_wmma_f32_16x16x32_f16(
            false, afr[i], false, bfr[j], (short)0, acc[i][j], false, false);

    // store prefetched B tile (and A, in the non-async fallback)
    if (have) {
#ifndef SGT_ASYNC
      _Float16* ad = &As[cur ^ 1][aRow * APAD + aCol];
      *(v8h*)(ad)     = na0;
      *(v8h*)(ad + 8) = na1;
#endif
      sgt_storeBpair(&Bs[cur ^ 1][0], bC8, bKp, nb0, nb1);
    }
#ifdef SGT_ASYNC
    sgt_wait_async();
#endif
    __syncthreads();
    cur ^= 1;
  }

  // Epilogue: C/D layout: VGPR r, lanes0-15 -> M=r, lanes16-31 -> M=8+r
  #pragma unroll
  for (int i = 0; i < 2; ++i)
    #pragma unroll
    for (int j = 0; j < 4; ++j) {
      int col = nBase + colG * 64 + j * 16 + lr;
      if (col < Nc) {
        float bv = bias ? bias[col] : 0.0f;
        int rb = mBase + rowG * 32 + i * 16 + hi * 8;
        #pragma unroll
        for (int r = 0; r < 8; ++r)
          Cmat[(size_t)(rb + r) * Nc + col] = acc[i][j][r] + bv;
      }
    }
}

// ---------------------------------------------------------------------------
// LayerNorm(D=512)+ReLU on y[N,D]; write identical planes h[t=0], h[t=1]
// ---------------------------------------------------------------------------
__global__ __launch_bounds__(256)
void sgt_ln_relu(const float* __restrict__ y, const float* __restrict__ g,
                 const float* __restrict__ b, float* __restrict__ h) {
  __shared__ float bs[256], bss[256];
  int row = blockIdx.x, tid = threadIdx.x;
  const float* yr = y + (size_t)row * DDIM;
  float s = 0.f, ss = 0.f;
  for (int c = tid; c < DDIM; c += 256) { float v = yr[c]; s += v; ss += v * v; }
  bs[tid] = s; bss[tid] = ss; __syncthreads();
  for (int off = 128; off; off >>= 1) {
    if (tid < off) { bs[tid] += bs[tid + off]; bss[tid] += bss[tid + off]; }
    __syncthreads();
  }
  float mean = bs[0] * (1.0f / DDIM);
  float var  = bss[0] * (1.0f / DDIM) - mean * mean;
  float rs   = rsqrtf(var + EPSV);
  for (int c = tid; c < DDIM; c += 256) {
    float v = fmaxf((yr[c] - mean) * rs * g[c] + b[c], 0.0f);
    h[(size_t)row * DDIM + c] = v;
    h[(size_t)N_NODES * DDIM + (size_t)row * DDIM + c] = v;
  }
}

// ---------------------------------------------------------------------------
// Per-column sum / sumsq over M rows (atomics into sums[0..Cc-1], sums[Cc..])
// ---------------------------------------------------------------------------
__global__ __launch_bounds__(256)
void sgt_col_stats(const float* __restrict__ y, float* __restrict__ sums, int M, int Cc) {
  float s0[4] = {0,0,0,0}, s1[4] = {0,0,0,0};
  int tid = threadIdx.x;
  for (int r = blockIdx.x; r < M; r += gridDim.x) {
    const float* row = y + (size_t)r * Cc;
    #pragma unroll
    for (int i = 0; i < 4; ++i) {
      int c = tid + i * 256;
      if (c < Cc) { float v = row[c]; s0[i] += v; s1[i] += v * v; }
    }
  }
  #pragma unroll
  for (int i = 0; i < 4; ++i) {
    int c = tid + i * 256;
    if (c < Cc) { atomicAdd(&sums[c], s0[i]); atomicAdd(&sums[Cc + c], s1[i]); }
  }
}

__global__ void sgt_bn_finalize(const float* __restrict__ sums, const float* __restrict__ g,
                                const float* __restrict__ b, float* __restrict__ sb,
                                int Cc, float invM) {
  int c = blockIdx.x * blockDim.x + threadIdx.x;
  if (c >= Cc) return;
  float mean = sums[c] * invM;
  float var  = sums[Cc + c] * invM - mean * mean;
  float sc = g[c] * rsqrtf(var + EPSV);
  sb[c] = sc;
  sb[Cc + c] = b[c] - mean * sc;
}

// ---------------------------------------------------------------------------
// BN(scale/bias) + LIF over T=2 -> f16 spikes
// ---------------------------------------------------------------------------
__global__ void sgt_bn_lif(const float* __restrict__ y, const float* __restrict__ sb,
                           _Float16* __restrict__ s, int plane, int Cc) {
  int i = blockIdx.x * blockDim.x + threadIdx.x;
  int st = gridDim.x * blockDim.x;
  for (; i < plane; i += st) {
    int c = i % Cc;
    float sc = sb[c], bi = sb[Cc + c];
    float x0 = sc * y[i] + bi;
    float x1 = sc * y[plane + i] + bi;
    float v = 0.5f * x0;
    float s0 = (v >= 1.0f) ? 1.0f : 0.0f;
    v *= (1.0f - s0);
    v = 0.5f * (v + x1);
    float s1 = (v >= 1.0f) ? 1.0f : 0.0f;
    s[i] = (_Float16)s0;
    s[plane + i] = (_Float16)s1;
  }
}

__global__ void sgt_lif(const float* __restrict__ y, _Float16* __restrict__ s, int plane) {
  int i = blockIdx.x * blockDim.x + threadIdx.x;
  int st = gridDim.x * blockDim.x;
  for (; i < plane; i += st) {
    float x0 = y[i], x1 = y[plane + i];
    float v = 0.5f * x0;
    float s0 = (v >= 1.0f) ? 1.0f : 0.0f;
    v *= (1.0f - s0);
    v = 0.5f * (v + x1);
    float s1 = (v >= 1.0f) ? 1.0f : 0.0f;
    s[i] = (_Float16)s0;
    s[plane + i] = (_Float16)s1;
  }
}

// h += scale*y + bias (both timesteps)
__global__ void sgt_bn_addres(const float* __restrict__ y, const float* __restrict__ sb,
                              float* __restrict__ h, int total, int Cc) {
  int i = blockIdx.x * blockDim.x + threadIdx.x;
  int st = gridDim.x * blockDim.x;
  for (; i < total; i += st) {
    int c = i % Cc;
    h[i] += sb[c] * y[i] + sb[Cc + c];
  }
}

// ---------------------------------------------------------------------------
// kv[t,h,d,e] = sum_n k[t,n,h*64+d] * v[t,n,h*64+e]   (spikes -> skip zeros)
// ---------------------------------------------------------------------------
__global__ __launch_bounds__(256)
void sgt_kv(const _Float16* __restrict__ kk, const _Float16* __restrict__ vv,
            float* __restrict__ kv, int chunk) {
  int th = blockIdx.x;
  int t = th >> 3, hd = th & 7;
  int n0 = blockIdx.y * chunk;
  int tid = threadIdx.x;
  int d = tid >> 2, e0 = (tid & 3) * 16;
  size_t base = (size_t)t * N_NODES * DDIM + (size_t)hd * 64;
  __shared__ _Float16 kb[8][64], vb[8][64];
  float acc[16];
  #pragma unroll
  for (int j = 0; j < 16; ++j) acc[j] = 0.f;
  for (int n = n0; n < n0 + chunk; n += 8) {
    int j0 = tid * 2;
    int r0 = j0 >> 6, c0 = j0 & 63;
    size_t rowoff = base + (size_t)(n + r0) * DDIM + c0;
    kb[r0][c0]     = kk[rowoff];
    kb[r0][c0 + 1] = kk[rowoff + 1];
    vb[r0][c0]     = vv[rowoff];
    vb[r0][c0 + 1] = vv[rowoff + 1];
    __syncthreads();
    #pragma unroll
    for (int r = 0; r < 8; ++r) {
      if ((float)kb[r][d] != 0.f) {
        #pragma unroll
        for (int j = 0; j < 16; ++j) acc[j] += (float)vb[r][e0 + j];
      }
    }
    __syncthreads();
  }
  float* dst = kv + (((size_t)(t * HHEADS + hd)) * 64 + d) * 64 + e0;
  #pragma unroll
  for (int j = 0; j < 16; ++j) atomicAdd(&dst[j], acc[j]);
}

// o[t,n,h*64+e] = SCALE * sum_d q[t,n,h*64+d] * kv[t,h,d,e]
__global__ __launch_bounds__(256)
void sgt_attn_out(const _Float16* __restrict__ q, const float* __restrict__ kv,
                  float* __restrict__ o, int rowsPB) {
  int th = blockIdx.x;
  int t = th >> 3, hd = th & 7;
  __shared__ float kvs[64 * 64];
  const float* src = kv + (size_t)(t * HHEADS + hd) * 4096;
  for (int i = threadIdx.x; i < 4096; i += 256) kvs[i] = src[i];
  __syncthreads();
  int nEnd = blockIdx.y * rowsPB + rowsPB;
  for (int n = blockIdx.y * rowsPB + threadIdx.x; n < nEnd; n += 256) {
    const _Float16* qr = q + (size_t)t * N_NODES * DDIM + (size_t)n * DDIM + hd * 64;
    float outv[64];
    #pragma unroll
    for (int e = 0; e < 64; ++e) outv[e] = 0.f;
    for (int dd = 0; dd < 64; ++dd) {
      if ((float)qr[dd] != 0.f) {
        const float* kr = &kvs[dd * 64];
        #pragma unroll
        for (int e = 0; e < 64; ++e) outv[e] += kr[e];
      }
    }
    float* orow = o + (size_t)t * N_NODES * DDIM + (size_t)n * DDIM + hd * 64;
    #pragma unroll
    for (int e = 0; e < 64; ++e) orow[e] = outv[e] * 0.125f;
  }
}

__global__ void sgt_head_mean(const float* __restrict__ y, float* __restrict__ x1, int plane) {
  int i = blockIdx.x * blockDim.x + threadIdx.x;
  int st = gridDim.x * blockDim.x;
  for (; i < plane; i += st) x1[i] = 0.5f * (y[i] + y[plane + i]);
}

// ---------------------------------------------------------------------------
// GNN branch helpers
// ---------------------------------------------------------------------------
__global__ void sgt_bn_relu(const float* __restrict__ y, const float* __restrict__ sb,
                            const float* __restrict__ g0, float* __restrict__ out,
                            int total, int Cc) {
  int i = blockIdx.x * blockDim.x + threadIdx.x;
  int st = gridDim.x * blockDim.x;
  for (; i < total; i += st) {
    int c = i % Cc;
    float v = fmaxf(sb[c] * y[i] + sb[Cc + c], 0.0f);
    if (g0) v += g0[i];
    out[i] = v;
  }
}

__global__ void sgt_deg(const int* __restrict__ col, float* __restrict__ deg, int E) {
  int i = blockIdx.x * blockDim.x + threadIdx.x;
  int st = gridDim.x * blockDim.x;
  for (; i < E; i += st) atomicAdd(&deg[col[i]], 1.0f);
}

__global__ void sgt_dinv(float* __restrict__ deg, int n) {
  int i = blockIdx.x * blockDim.x + threadIdx.x;
  int st = gridDim.x * blockDim.x;
  for (; i < n; i += st) { float v = deg[i]; deg[i] = (v > 0.f) ? rsqrtf(v) : 0.f; }
}

__global__ void sgt_gcn_scatter(const int* __restrict__ row, const int* __restrict__ col,
                                const float* __restrict__ dinv, const float* __restrict__ g,
                                float* __restrict__ agg, int E) {
  int e = blockIdx.x * blockDim.x + threadIdx.x;
  int st = gridDim.x * blockDim.x;
  for (; e < E; e += st) {
    int rr = row[e], cc = col[e];
    float val = dinv[rr] * dinv[cc];
    const float* gr = g + (size_t)rr * CDIM;
    float* ar = agg + (size_t)cc * CDIM;
    #pragma unroll
    for (int f = 0; f < CDIM; ++f) atomicAdd(&ar[f], val * gr[f]);
  }
}

__global__ __launch_bounds__(256)
void sgt_small_gemm(const float* __restrict__ Ain, const float* __restrict__ W,
                    const float* __restrict__ b, float* __restrict__ out, int M) {
  __shared__ float Ws[CDIM * CDIM];
  for (int i = threadIdx.x; i < CDIM * CDIM; i += 256) Ws[i] = W[i];
  __syncthreads();
  int idx = blockIdx.x * 256 + threadIdx.x;
  int st = gridDim.x * 256;
  int total = M * CDIM;
  for (; idx < total; idx += st) {
    int n = idx / CDIM, c = idx - n * CDIM;
    const float* ar = Ain + (size_t)n * CDIM;
    float s = b ? b[c] : 0.f;
    #pragma unroll
    for (int k = 0; k < CDIM; ++k) s += ar[k] * Ws[k * CDIM + c];
    out[idx] = s;
  }
}

__global__ __launch_bounds__(256)
void sgt_final(const float* __restrict__ g, const float* __restrict__ x1,
               const float* __restrict__ W, const float* __restrict__ b,
               float* __restrict__ out, int M) {
  __shared__ float Ws[CDIM * CDIM];
  for (int i = threadIdx.x; i < CDIM * CDIM; i += 256) Ws[i] = W[i];
  __syncthreads();
  int idx = blockIdx.x * 256 + threadIdx.x;
  int st = gridDim.x * 256;
  int total = M * CDIM;
  for (; idx < total; idx += st) {
    int n = idx / CDIM, c = idx - n * CDIM;
    const float* gr = g + (size_t)n * CDIM;
    const float* xr = x1 + (size_t)n * CDIM;
    float s = b[c];
    #pragma unroll
    for (int k = 0; k < CDIM; ++k) s += (0.8f * gr[k] + 0.2f * xr[k]) * Ws[k * CDIM + c];
    out[idx] = s;
  }
}

// ---------------------------------------------------------------------------
// Host orchestration
// ---------------------------------------------------------------------------
static inline int ewblocks(long long n) {
  long long b = (n + 255) / 256;
  if (b > 32768) b = 32768;
  return (int)b;
}

extern "C" void kernel_launch(void* const* d_in, const int* in_sizes, int n_in,
                              void* d_out, int out_size, void* d_ws, size_t ws_size,
                              hipStream_t stream) {
  (void)in_sizes; (void)n_in; (void)out_size; (void)ws_size;
  const float* x     = (const float*)d_in[0];
  const float* fc0w  = (const float*)d_in[1];
  const float* fc0b  = (const float*)d_in[2];
  const float* lng   = (const float*)d_in[3];
  const float* lnb   = (const float*)d_in[4];
  const float* wq    = (const float*)d_in[5];
  const float* wk    = (const float*)d_in[6];
  const float* wv    = (const float*)d_in[7];
  const float* wo    = (const float*)d_in[8];
  const float* bnqg  = (const float*)d_in[9];
  const float* bnqb  = (const float*)d_in[10];
  const float* bnkg  = (const float*)d_in[11];
  const float* bnkb  = (const float*)d_in[12];
  const float* bnvg  = (const float*)d_in[13];
  const float* bnvb  = (const float*)d_in[14];
  const float* bnog  = (const float*)d_in[15];
  const float* bnob  = (const float*)d_in[16];
  const float* w1    = (const float*)d_in[17];
  const float* bn1g  = (const float*)d_in[18];
  const float* bn1b  = (const float*)d_in[19];
  const float* w2    = (const float*)d_in[20];
  const float* bn2g  = (const float*)d_in[21];
  const float* bn2b  = (const float*)d_in[22];
  const float* headw = (const float*)d_in[23];
  const float* headb = (const float*)d_in[24];
  const float* gfcw  = (const float*)d_in[25];
  const float* gfcb  = (const float*)d_in[26];
  const float* gbn0g = (const float*)d_in[27];
  const float* gbn0b = (const float*)d_in[28];
  const float* gconvw= (const float*)d_in[29];
  const float* gconvb= (const float*)d_in[30];
  const float* gbng  = (const float*)d_in[31];
  const float* gbnb  = (const float*)d_in[32];
  const float* fcw   = (const float*)d_in[33];
  const float* fcb   = (const float*)d_in[34];
  const int*   eidx  = (const int*)d_in[35];
  const int* erow = eidx;
  const int* ecol = eidx + EEDGES;

  // ---- workspace carve (≈370 MB) ----
  char* wsp = (char*)d_ws;
  auto alloc = [&](size_t bytes) -> void* {
    void* p = (void*)wsp;
    wsp += (bytes + 255) & ~(size_t)255;
    return p;
  };
  const size_t ND  = (size_t)N_NODES * DDIM;     // 8.4M
  const size_t NDH = (size_t)N_NODES * DHDIM;    // 16.8M
  float*    hbuf  = (float*)alloc(TT * ND * 4);            // 64 MB
  float*    yscr  = (float*)alloc(TT * NDH * 4);           // 128 MB
  _Float16* s16   = (_Float16*)alloc(TT * ND * 2);         // 32 MB
  _Float16* qkv16 = (_Float16*)alloc(3 * TT * ND * 2);     // 96 MB
  _Float16* q16 = qkv16;
  _Float16* k16 = qkv16 + TT * ND;
  _Float16* v16 = qkv16 + 2 * TT * ND;
  _Float16* m16 = qkv16;                                   // alias q+k (2N x DH)
  _Float16* x16   = (_Float16*)alloc((size_t)N_NODES * FDIM * 2);     // 16 MB
  _Float16* fc016 = (_Float16*)alloc((size_t)FDIM * DDIM * 2);
  _Float16* wq16  = (_Float16*)alloc((size_t)LLAYERS * DDIM * DDIM * 2);
  _Float16* wk16  = (_Float16*)alloc((size_t)LLAYERS * DDIM * DDIM * 2);
  _Float16* wv16  = (_Float16*)alloc((size_t)LLAYERS * DDIM * DDIM * 2);
  _Float16* wo16  = (_Float16*)alloc((size_t)LLAYERS * DDIM * DDIM * 2);
  _Float16* w116  = (_Float16*)alloc((size_t)LLAYERS * DDIM * DHDIM * 2);
  _Float16* w216  = (_Float16*)alloc((size_t)LLAYERS * DHDIM * DDIM * 2);
  _Float16* head16= (_Float16*)alloc((size_t)DDIM * CDIM * 2);
  _Float16* gfc16 = (_Float16*)alloc((size_t)FDIM * CDIM * 2);
  float* sums  = (float*)alloc(2 * DHDIM * 4);
  float* sb    = (float*)alloc(2 * DHDIM * 4);
  float* kvbuf = (float*)alloc((size_t)TT * HHEADS * 64 * 64 * 4);
  float* x1    = (float*)alloc((size_t)N_NODES * CDIM * 4);
  float* g0    = (float*)alloc((size_t)N_NODES * CDIM * 4);
  float* gbuf  = (float*)alloc((size_t)N_NODES * CDIM * 4);
  float* agg   = (float*)alloc((size_t)N_NODES * CDIM * 4);
  float* gy    = (float*)alloc((size_t)N_NODES * CDIM * 4);
  float* deg   = (float*)alloc((size_t)N_NODES * 4);

  auto cvt = [&](const float* src, _Float16* dst, long long n) {
    sgt_cvt_f16<<<ewblocks(n), 256, 0, stream>>>(src, dst, (int)n);
  };
  auto gemm = [&](const _Float16* A, const _Float16* B, float* C,
                  const float* bias, int M, int K, int Nc) {
    dim3 grid(M / GBM, (Nc + GBN - 1) / GBN);
    sgt_gemm<<<grid, 256, 0, stream>>>(A, B, C, bias, M, K, Nc);
  };
  auto stats = [&](const float* y, int Mr, int Cc, const float* g, const float* b) {
    (void)hipMemsetAsync(sums, 0, 2 * Cc * sizeof(float), stream);
    sgt_col_stats<<<512, 256, 0, stream>>>(y, sums, Mr, Cc);
    sgt_bn_finalize<<<(Cc + 255) / 256, 256, 0, stream>>>(sums, g, b, sb, Cc, 1.0f / Mr);
  };

  const int M2N = TT * N_NODES;   // 32768

  // ---- weight / input conversions to f16 ----
  cvt(x, x16, (long long)N_NODES * FDIM);
  cvt(fc0w, fc016, (long long)FDIM * DDIM);
  cvt(wq, wq16, (long long)LLAYERS * DDIM * DDIM);
  cvt(wk, wk16, (long long)LLAYERS * DDIM * DDIM);
  cvt(wv, wv16, (long long)LLAYERS * DDIM * DDIM);
  cvt(wo, wo16, (long long)LLAYERS * DDIM * DDIM);
  cvt(w1, w116, (long long)LLAYERS * DDIM * DHDIM);
  cvt(w2, w216, (long long)LLAYERS * DHDIM * DDIM);
  cvt(headw, head16, (long long)DDIM * CDIM);
  cvt(gfcw, gfc16, (long long)FDIM * CDIM);

  // ---- fc0 + LN + ReLU -> h (both timesteps identical) ----
  gemm(x16, fc016, yscr, fc0b, N_NODES, FDIM, DDIM);
  sgt_ln_relu<<<N_NODES, 256, 0, stream>>>(yscr, lng, lnb, hbuf);

  // ---- transformer layers ----
  for (int l = 0; l < LLAYERS; ++l) {
    sgt_lif<<<ewblocks(ND), 256, 0, stream>>>(hbuf, s16, (int)ND);
    // q
    gemm(s16, wq16 + (size_t)l * DDIM * DDIM, yscr, nullptr, M2N, DDIM, DDIM);
    stats(yscr, M2N, DDIM, bnqg + l * DDIM, bnqb + l * DDIM);
    sgt_bn_lif<<<ewblocks(ND), 256, 0, stream>>>(yscr, sb, q16, (int)ND, DDIM);
    // k
    gemm(s16, wk16 + (size_t)l * DDIM * DDIM, yscr, nullptr, M2N, DDIM, DDIM);
    stats(yscr, M2N, DDIM, bnkg + l * DDIM, bnkb + l * DDIM);
    sgt_bn_lif<<<ewblocks(ND), 256, 0, stream>>>(yscr, sb, k16, (int)ND, DDIM);
    // v
    gemm(s16, wv16 + (size_t)l * DDIM * DDIM, yscr, nullptr, M2N, DDIM, DDIM);
    stats(yscr, M2N, DDIM, bnvg + l * DDIM, bnvb + l * DDIM);
    sgt_bn_lif<<<ewblocks(ND), 256, 0, stream>>>(yscr, sb, v16, (int)ND, DDIM);
    // attention: kv then o = q@kv * SCALE
    (void)hipMemsetAsync(kvbuf, 0, (size_t)TT * HHEADS * 64 * 64 * 4, stream);
    sgt_kv<<<dim3(TT * HHEADS, 8), 256, 0, stream>>>(k16, v16, kvbuf, N_NODES / 8);
    sgt_attn_out<<<dim3(TT * HHEADS, 32), 256, 0, stream>>>(q16, kvbuf, yscr, N_NODES / 32);
    // o = bn(lif(o) @ wo); h += o
    sgt_lif<<<ewblocks(ND), 256, 0, stream>>>(yscr, s16, (int)ND);
    gemm(s16, wo16 + (size_t)l * DDIM * DDIM, yscr, nullptr, M2N, DDIM, DDIM);
    stats(yscr, M2N, DDIM, bnog + l * DDIM, bnob + l * DDIM);
    sgt_bn_addres<<<ewblocks(2 * ND), 256, 0, stream>>>(yscr, sb, hbuf, (int)(2 * ND), DDIM);
    // MLP
    sgt_lif<<<ewblocks(ND), 256, 0, stream>>>(hbuf, s16, (int)ND);
    gemm(s16, w116 + (size_t)l * DDIM * DHDIM, yscr, nullptr, M2N, DDIM, DHDIM);
    stats(yscr, M2N, DHDIM, bn1g + l * DHDIM, bn1b + l * DHDIM);
    sgt_bn_lif<<<ewblocks(NDH), 256, 0, stream>>>(yscr, sb, m16, (int)NDH, DHDIM);
    gemm(m16, w216 + (size_t)l * DHDIM * DDIM, yscr, nullptr, M2N, DHDIM, DDIM);
    stats(yscr, M2N, DDIM, bn2g + l * DDIM, bn2b + l * DDIM);
    sgt_bn_addres<<<ewblocks(2 * ND), 256, 0, stream>>>(yscr, sb, hbuf, (int)(2 * ND), DDIM);
  }

  // ---- head ----
  sgt_lif<<<ewblocks(ND), 256, 0, stream>>>(hbuf, s16, (int)ND);
  gemm(s16, head16, yscr, headb, M2N, DDIM, CDIM);
  sgt_head_mean<<<ewblocks((long long)N_NODES * CDIM), 256, 0, stream>>>(
      yscr, x1, N_NODES * CDIM);

  // ---- GNN branch ----
  gemm(x16, gfc16, gy, gfcb, N_NODES, FDIM, CDIM);
  stats(gy, N_NODES, CDIM, gbn0g, gbn0b);
  sgt_bn_relu<<<ewblocks((long long)N_NODES * CDIM), 256, 0, stream>>>(
      gy, sb, nullptr, g0, N_NODES * CDIM, CDIM);
  (void)hipMemcpyAsync(gbuf, g0, (size_t)N_NODES * CDIM * 4, hipMemcpyDeviceToDevice, stream);

  (void)hipMemsetAsync(deg, 0, (size_t)N_NODES * 4, stream);
  sgt_deg<<<ewblocks(EEDGES), 256, 0, stream>>>(ecol, deg, EEDGES);
  sgt_dinv<<<ewblocks(N_NODES), 256, 0, stream>>>(deg, N_NODES);

  for (int i = 0; i < 2; ++i) {
    (void)hipMemsetAsync(agg, 0, (size_t)N_NODES * CDIM * 4, stream);
    sgt_gcn_scatter<<<ewblocks(EEDGES), 256, 0, stream>>>(erow, ecol, deg, gbuf, agg, EEDGES);
    sgt_small_gemm<<<ewblocks((long long)N_NODES * CDIM), 256, 0, stream>>>(
        agg, gconvw + i * CDIM * CDIM, gconvb + i * CDIM, gy, N_NODES);
    stats(gy, N_NODES, CDIM, gbng + i * CDIM, gbnb + i * CDIM);
    sgt_bn_relu<<<ewblocks((long long)N_NODES * CDIM), 256, 0, stream>>>(
        gy, sb, g0, gbuf, N_NODES * CDIM, CDIM);
  }

  // ---- combine + final projection ----
  sgt_final<<<ewblocks((long long)N_NODES * CDIM), 256, 0, stream>>>(
      gbuf, x1, fcw, fcb, (float*)d_out, N_NODES);
}
